// RWKV_14078902796624
// MI455X (gfx1250) — matrix operand order repaired and checked by reading, not compiled, not run on hardware
//
#include <hip/hip_runtime.h>
#include <math.h>

// ---------------- problem constants (fixed by the reference) ----------------
#define BB   2
#define TT   2048
#define EE   1024
#define AA   1024
#define FF   4096
#define NBLK 4
#define VV   32000
#define MTOK (BB*TT)          // 4096 token rows

typedef __bf16 bf16x16 __attribute__((ext_vector_type(16)));
typedef float  v8f     __attribute__((ext_vector_type(8)));

#define LDS_STRIDE 40         // bf16 elems per LDS row: 80B, keeps 16B alignment

// ---------------------------------------------------------------------------
// CDNA5 async global->LDS helpers (ASYNCcnt-tracked; no VGPR round trip)
// ---------------------------------------------------------------------------
__device__ __forceinline__ void async_ld16(unsigned ldsOff, const __bf16* g)
{
  asm volatile("global_load_async_to_lds_b128 %0, %1, off"
               :: "v"(ldsOff), "v"(g) : "memory");
}
__device__ __forceinline__ void wait_async8()
{ asm volatile("s_wait_asynccnt 0x8" ::: "memory"); }
__device__ __forceinline__ void wait_async0()
{ asm volatile("s_wait_asynccnt 0x0" ::: "memory"); }

// ---------------------------------------------------------------------------
// Per-RWKV-block constants: tm_k, tm_v, tm_r, cm_k [E], w=-exp(decay), u [A]
// ---------------------------------------------------------------------------
__global__ __launch_bounds__(256) void init_consts_kernel(float* __restrict__ cst)
{
  const int blk = blockIdx.x;
  float* base = cst + blk * (4*EE + 2*AA);
  const double r0 = blk / (double)(NBLK - 1);
  const double r1 = 1.0 - blk / (double)NBLK;
  for (int c = threadIdx.x; c < EE; c += blockDim.x) {
    double ddd = c / (double)EE;
    double tk  = pow(ddd, r1);
    base[0*EE + c] = (float)tk;
    base[1*EE + c] = (float)(tk + 0.3 * r0);
    base[2*EE + c] = (float)pow(ddd, 0.5 * r1);
    base[3*EE + c] = (float)tk;                       // cm_k
    double ds = -5.0 + 8.0 * pow(c / (double)(AA - 1), 0.7 + 1.3 * r0);
    base[4*EE + c] = -__expf((float)ds);              // w = -exp(time_decay)
    float zig = (float)(((c + 1) % 3) - 1) * 0.5f;
    base[4*EE + AA + c] = logf(0.3f) + zig;           // u = time_first
  }
}

// ---------------------------------------------------------------------------
// f32 -> bf16 bulk conversion (row-major copy; used for the head activation)
// ---------------------------------------------------------------------------
__global__ __launch_bounds__(256) void cvt_kernel(__bf16* __restrict__ dst,
                                                  const float* __restrict__ src,
                                                  int n4)
{
  const int i = blockIdx.x * 256 + threadIdx.x;
  if (i >= n4) return;
  const float4 f = ((const float4*)src)[i];
  __bf16* p = dst + (size_t)i * 4;
  p[0] = (__bf16)f.x; p[1] = (__bf16)f.y; p[2] = (__bf16)f.z; p[3] = (__bf16)f.w;
}

// ---------------------------------------------------------------------------
// Transposing f32 -> bf16 weight conversion: W[K][N] -> Wt[N][K].
// Makes the GEMM B-tile contiguous in K so it stages with pure b128 copies.
// ---------------------------------------------------------------------------
__global__ __launch_bounds__(256) void cvtT_kernel(__bf16* __restrict__ dst,
                                                   const float* __restrict__ src,
                                                   int K, int N)
{
  __shared__ float tile[32][33];
  const int tx = threadIdx.x & 31;
  const int ty = threadIdx.x >> 5;            // 0..7
  const int n0 = blockIdx.x * 32, k0 = blockIdx.y * 32;
  #pragma unroll
  for (int r = 0; r < 4; ++r)
    tile[ty + r*8][tx] = src[(size_t)(k0 + ty + r*8) * N + n0 + tx];
  __syncthreads();
  #pragma unroll
  for (int r = 0; r < 4; ++r)
    dst[(size_t)(n0 + ty + r*8) * K + k0 + tx] = (__bf16)tile[tx][ty + r*8];
}

// ---------------------------------------------------------------------------
// Embedding gather: h[row][:] = emb[x[row]][:]
// ---------------------------------------------------------------------------
__global__ __launch_bounds__(256) void embed_kernel(float* __restrict__ h,
                                                    const int* __restrict__ x,
                                                    const float* __restrict__ emb)
{
  const int row = blockIdx.x;
  const int tok = x[row];
  const float4* src = (const float4*)(emb + (size_t)tok * EE);
  float4*       dst = (float4*)(h + (size_t)row * EE);
  for (int i = threadIdx.x; i < EE/4; i += blockDim.x) dst[i] = src[i];
}

// ---------------------------------------------------------------------------
// LayerNorm over last dim (C = 1024), one block per row; safe in-place.
// ---------------------------------------------------------------------------
__global__ __launch_bounds__(256) void ln_kernel(float* __restrict__ out,
                                                 const float* __restrict__ in,
                                                 const float* __restrict__ g,
                                                 const float* __restrict__ b)
{
  const int row = blockIdx.x;
  const float* x = in + (size_t)row * EE;
  float s = 0.f, s2 = 0.f;
  for (int i = threadIdx.x; i < EE; i += 256) { float v = x[i]; s += v; s2 += v*v; }
  __shared__ float rs[8], rs2[8];
  for (int o = 16; o > 0; o >>= 1) { s += __shfl_xor(s, o, 32); s2 += __shfl_xor(s2, o, 32); }
  if ((threadIdx.x & 31) == 0) { rs[threadIdx.x >> 5] = s; rs2[threadIdx.x >> 5] = s2; }
  __syncthreads();
  s = 0.f; s2 = 0.f;
  #pragma unroll
  for (int i = 0; i < 8; ++i) { s += rs[i]; s2 += rs2[i]; }
  const float mean = s * (1.0f / EE);
  const float var  = s2 * (1.0f / EE) - mean * mean;
  const float inv  = rsqrtf(var + 1e-5f);
  float* o = out + (size_t)row * EE;
  for (int i = threadIdx.x; i < EE; i += 256)
    o[i] = (x[i] - mean) * inv * g[i] + b[i];
}

// ---------------------------------------------------------------------------
// Time-shift mixes -> bf16 outputs (GEMM A-operands are bf16)
// ---------------------------------------------------------------------------
__global__ __launch_bounds__(256) void mix3_kernel(__bf16* __restrict__ xk,
                                                   __bf16* __restrict__ xv,
                                                   __bf16* __restrict__ xr,
                                                   const float* __restrict__ x,
                                                   const float* __restrict__ mk,
                                                   const float* __restrict__ mv,
                                                   const float* __restrict__ mr)
{
  const int idx = blockIdx.x * 256 + threadIdx.x;
  if (idx >= MTOK * EE) return;
  const int c  = idx & (EE - 1);
  const int t  = (idx >> 10) & (TT - 1);
  const float xc = x[idx];
  const float xp = (t == 0) ? 0.f : x[idx - EE];
  const float a = mk[c], b = mv[c], d = mr[c];
  xk[idx] = (__bf16)(xc * a + xp * (1.f - a));
  xv[idx] = (__bf16)(xc * b + xp * (1.f - b));
  xr[idx] = (__bf16)(xc * d + xp * (1.f - d));
}

__global__ __launch_bounds__(256) void mix1_kernel(__bf16* __restrict__ xk,
                                                   const float* __restrict__ x,
                                                   const float* __restrict__ mk)
{
  const int idx = blockIdx.x * 256 + threadIdx.x;
  if (idx >= MTOK * EE) return;
  const int c  = idx & (EE - 1);
  const int t  = (idx >> 10) & (TT - 1);
  const float xc = x[idx];
  const float xp = (t == 0) ? 0.f : x[idx - EE];
  const float a = mk[c];
  xk[idx] = (__bf16)(xc * a + xp * (1.f - a));
}

// rw = sr * wkv -> bf16 (A-operand of the att-output GEMM)
__global__ __launch_bounds__(256) void ewmul_kernel(__bf16* __restrict__ o,
                                                    const float* __restrict__ a,
                                                    const float* __restrict__ b)
{
  const int idx = blockIdx.x * 256 + threadIdx.x;
  if (idx < MTOK * EE) o[idx] = (__bf16)(a[idx] * b[idx]);
}

// ---------------------------------------------------------------------------
// Numerically-stabilized WKV scan; one lane per (batch, channel).
// ---------------------------------------------------------------------------
__global__ __launch_bounds__(256) void wkv_kernel(float* __restrict__ y,
                                                  const float* __restrict__ k,
                                                  const float* __restrict__ v,
                                                  const float* __restrict__ wb,
                                                  const float* __restrict__ ub)
{
  const int idx = blockIdx.x * 256 + threadIdx.x;
  if (idx >= BB * AA) return;
  const int b = idx / AA, c = idx - b * AA;
  const float w = wb[c], u = ub[c];
  float aa = 0.f, bb = 0.f, pp = -1e38f;
  size_t p = (size_t)b * TT * AA + c;
  #pragma unroll 4
  for (int t = 0; t < TT; ++t, p += AA) {
    const float kk = k[p], vv = v[p];
    float ww = u + kk;
    float q  = fmaxf(pp, ww);
    float e1 = __expf(pp - q);
    float e2 = __expf(ww - q);
    y[p] = (e1 * aa + e2 * vv) / (e1 * bb + e2);
    ww = pp + w;
    q  = fmaxf(ww, kk);
    e1 = __expf(ww - q);
    e2 = __expf(kk - q);
    aa = e1 * aa + e2 * vv;
    bb = e1 * bb + e2;
    pp = q;
  }
}

// ---------------------------------------------------------------------------
// bf16 WMMA GEMM:  out[M,N] = epi(act[M,K] @ Wt[N,K]^T) (* mul) (+ res)
// 128 threads = 4 waves; block tile 128x128; K-step 32; wave tile 64x64
// (4x4 of v_wmma_f32_16x16x32_bf16 => 16 WMMAs per 16 ds_load_b128).
// Both tiles staged by global_load_async_to_lds_b128 into double-buffered LDS;
// next K-tile's copies overlap the current tile's WMMAs; tail is peeled so the
// steady-state loop is branch-free.
// EPI: 0 = none, 1 = sigmoid, 2 = relu^2.  OBF: store bf16 instead of f32.
// ---------------------------------------------------------------------------
__device__ __forceinline__ bf16x16 lds_frag(const __bf16* base, int row, int o0, int o1)
{
  union { bf16x16 v; uint4 q[2]; } u;
  u.q[0] = *(const uint4*)(base + row * LDS_STRIDE + o0);
  u.q[1] = *(const uint4*)(base + row * LDS_STRIDE + o1);
  return u.v;
}

template<int EPI, bool RESF, bool MULF, bool OBF>
__device__ __forceinline__ void store_tile(void* __restrict__ outv,
                                           const float* __restrict__ res,
                                           const float* __restrict__ mul,
                                           v8f acc, int gm0, int gn0, int N,
                                           int h, int lm)
{
  const int gn = gn0 + lm;
  #pragma unroll
  for (int r = 0; r < 8; ++r) {
    const int gm = gm0 + h * 8 + r;            // C/D layout: VGPR r -> M = r + 8*half
    const size_t idx = (size_t)gm * N + gn;
    float val = acc[r];
    if (EPI == 1) val = __builtin_amdgcn_rcpf(1.0f + __expf(-val));
    if (EPI == 2) { val = fmaxf(val, 0.f); val = val * val; }
    if (MULF) val *= mul[idx];
    if (RESF) val += res[idx];
    if (OBF) ((__bf16*)outv)[idx] = (__bf16)val;
    else     ((float*)outv)[idx]  = val;
  }
}

template<int EPI, bool RESF, bool MULF, bool OBF>
__global__ __launch_bounds__(128) void gemm_kernel(void* __restrict__ outv,
                                                   const __bf16* __restrict__ act,
                                                   const __bf16* __restrict__ wgtT,
                                                   int N, int K,
                                                   const float* __restrict__ res,
                                                   const float* __restrict__ mul)
{
  __shared__ __align__(16) __bf16 As[2][128 * LDS_STRIDE];   // [m][k]
  __shared__ __align__(16) __bf16 Bs[2][128 * LDS_STRIDE];   // [n][k]

  const int tid  = threadIdx.x;
  const int lane = tid & 31, wave = tid >> 5;
  const int wm = (wave >> 1) * 64;        // 0,64
  const int wn = (wave & 1)  * 64;        // 0,64
  const int h = lane >> 4, lm = lane & 15;
  const int mBase = blockIdx.y * 128, nBase = blockIdx.x * 128;

  v8f acc[4][4];
  #pragma unroll
  for (int i = 0; i < 4; ++i)
    #pragma unroll
    for (int j = 0; j < 4; ++j)
      acc[i][j] = (v8f){0,0,0,0,0,0,0,0};

  // 8 async b128 copies per thread per tile (4 A-chunks + 4 B-chunks)
  auto stage = [&](int b2, int k0) {
    #pragma unroll
    for (int j = 0; j < 4; ++j) {
      const int q = tid + j * 128;               // 0..511
      const int m = q >> 2, kk = (q & 3) << 3;   // m:0..127, kk:0,8,16,24
      async_ld16((unsigned)(size_t)&As[b2][m * LDS_STRIDE + kk],
                 act  + (size_t)(mBase + m) * K + k0 + kk);
      async_ld16((unsigned)(size_t)&Bs[b2][m * LDS_STRIDE + kk],
                 wgtT + (size_t)(nBase + m) * K + k0 + kk);
    }
  };

  // fragments per ISA 16-bit layouts:
  //  A 16x32: v0-3 = K[h*8 .. +7], v4-7 = K[16+h*8 .. +7], row = lane%16
  //  B 32x16: v0-3 = K[h*8 .. +7], v4-7 = K[16+h*8 .. +7], col = lane%16
  auto compute = [&](int b2) {
    const __bf16* As_ = As[b2];
    const __bf16* Bs_ = Bs[b2];
    bf16x16 afr[4], bfr[4];
    #pragma unroll
    for (int i = 0; i < 4; ++i)
      afr[i] = lds_frag(As_, wm + i * 16 + lm, h * 8, 16 + h * 8);
    #pragma unroll
    for (int j = 0; j < 4; ++j)
      bfr[j] = lds_frag(Bs_, wn + j * 16 + lm, h * 8, 16 + h * 8);
    #pragma unroll
    for (int i = 0; i < 4; ++i)
      #pragma unroll
      for (int j = 0; j < 4; ++j)
        acc[i][j] = __builtin_amdgcn_wmma_f32_16x16x32_bf16(
            false, afr[i], false, bfr[j], (short)0, acc[i][j], false, false);
  };

  stage(0, 0);
  int buf = 0;
  // branch-free steady state: always one tile in flight
  for (int k0 = 0; k0 + 32 < K; k0 += 32) {
    stage(buf ^ 1, k0 + 32);
    wait_async8();                 // previous buffer's 8 copies complete
    __syncthreads();               // visible to all waves
    compute(buf);
    __syncthreads();               // all reads done before overwrite
    buf ^= 1;
  }
  wait_async0();                   // peeled tail
  __syncthreads();
  compute(buf);

  #pragma unroll
  for (int i = 0; i < 4; ++i)
    #pragma unroll
    for (int j = 0; j < 4; ++j)
      store_tile<EPI, RESF, MULF, OBF>(outv, res, mul, acc[i][j],
                                       mBase + wm + i * 16,
                                       nBase + wn + j * 16, N, h, lm);
}

// dispatch helper: only the combos actually used get instantiated
enum { G_PLAIN = 0, G_SIG, G_RELUSQ_BF, G_RES, G_RESMUL };
static void launch_gemm(int kind, void* out, const __bf16* act, const __bf16* wT,
                        int N, int K, const float* res, const float* mul,
                        hipStream_t s)
{
  dim3 grid(N / 128, MTOK / 128), block(128);
  switch (kind) {
    case G_SIG:       gemm_kernel<1,false,false,false><<<grid,block,0,s>>>(out, act, wT, N, K, nullptr, nullptr); break;
    case G_RELUSQ_BF: gemm_kernel<2,false,false,true ><<<grid,block,0,s>>>(out, act, wT, N, K, nullptr, nullptr); break;
    case G_RES:       gemm_kernel<0,true ,false,false><<<grid,block,0,s>>>(out, act, wT, N, K, res,     nullptr); break;
    case G_RESMUL:    gemm_kernel<0,true ,true ,false><<<grid,block,0,s>>>(out, act, wT, N, K, res,     mul);     break;
    default:          gemm_kernel<0,false,false,false><<<grid,block,0,s>>>(out, act, wT, N, K, nullptr, nullptr); break;
  }
}

// ---------------------------------------------------------------------------
// Host orchestration
// ---------------------------------------------------------------------------
struct Blk {
  const float *ln1g, *ln1b, *ln2g, *ln2b;
  const float *attk, *attv, *attr, *atto;
  const float *ffnk, *ffnr, *ffnv;
  const float *ln0g, *ln0b;
};

static void cvtT(__bf16* dst, const float* src, int K, int N, hipStream_t s)
{
  dim3 grid(N / 32, K / 32), block(256);
  cvtT_kernel<<<grid, block, 0, s>>>(dst, src, K, N);
}

extern "C" void kernel_launch(void* const* d_in, const int* in_sizes, int n_in,
                              void* d_out, int out_size, void* d_ws, size_t ws_size,
                              hipStream_t stream)
{
  (void)in_sizes; (void)n_in; (void)out_size; (void)ws_size;

  // ---- inputs in setup_inputs() dict insertion order ----
  int cur = 0;
  const int*   x   = (const int*)d_in[cur++];
  const float* emb = (const float*)d_in[cur++];
  Blk blk[NBLK];
  for (int i = 0; i < NBLK; ++i) {
    blk[i].ln1g = (const float*)d_in[cur++];
    blk[i].ln1b = (const float*)d_in[cur++];
    blk[i].ln2g = (const float*)d_in[cur++];
    blk[i].ln2b = (const float*)d_in[cur++];
    blk[i].attk = (const float*)d_in[cur++];
    blk[i].attv = (const float*)d_in[cur++];
    blk[i].attr = (const float*)d_in[cur++];
    blk[i].atto = (const float*)d_in[cur++];
    blk[i].ffnk = (const float*)d_in[cur++];
    blk[i].ffnr = (const float*)d_in[cur++];
    blk[i].ffnv = (const float*)d_in[cur++];
    if (i == 0) { blk[i].ln0g = (const float*)d_in[cur++];
                  blk[i].ln0b = (const float*)d_in[cur++]; }
    else        { blk[i].ln0g = nullptr; blk[i].ln0b = nullptr; }
  }
  const float* ln_out_g = (const float*)d_in[cur++];
  const float* ln_out_b = (const float*)d_in[cur++];
  const float* head     = (const float*)d_in[cur++];

  // ---- workspace carve-up ----
  const size_t S = (size_t)MTOK * EE;           // 4M elements
  float* ws  = (float*)d_ws;
  float* CST = ws;                              // 4 * 6144 floats
  float* H   = CST + NBLK * (4*EE + 2*AA);
  float* XLN = H   + S;
  float* K4  = XLN + S;                         // k
  float* V5  = K4  + S;                         // v
  float* R6  = V5  + S;                         // sigmoid(r)
  float* WK  = R6  + S;                         // wkv output
  __bf16* bws = (__bf16*)(WK + S);
  __bf16* B1  = bws;                            // xk / rw / cm-x   (bf16)
  __bf16* B2  = B1 + S;                         // xv
  __bf16* B3  = B2 + S;                         // xr
  __bf16* XLB = B3 + S;                         // bf16 copy of final LN
  __bf16* FKB = XLB + S;                        // relu(k@Wk)^2 [MTOK, FF]
  __bf16* wb  = FKB + (size_t)MTOK * FF;        // bf16 transposed weights

  struct WB { __bf16 *attk, *attv, *attr, *atto, *ffnk, *ffnr, *ffnv; } w16[NBLK];
  const size_t SQ = (size_t)EE * AA;            // 1M
  const size_t SF = (size_t)EE * FF;            // 4M
  for (int i = 0; i < NBLK; ++i) {
    w16[i].attk = wb; wb += SQ;
    w16[i].attv = wb; wb += SQ;
    w16[i].attr = wb; wb += SQ;
    w16[i].atto = wb; wb += SQ;
    w16[i].ffnk = wb; wb += SF;
    w16[i].ffnr = wb; wb += SQ;
    w16[i].ffnv = wb; wb += SF;
  }
  __bf16* headB = wb; wb += (size_t)EE * VV;

  // ---- one-shot transposed weight conversion (bandwidth-trivial) ----
  for (int i = 0; i < NBLK; ++i) {
    cvtT(w16[i].attk, blk[i].attk, EE, AA, stream);
    cvtT(w16[i].attv, blk[i].attv, EE, AA, stream);
    cvtT(w16[i].attr, blk[i].attr, EE, AA, stream);
    cvtT(w16[i].atto, blk[i].atto, AA, EE, stream);
    cvtT(w16[i].ffnk, blk[i].ffnk, EE, FF, stream);
    cvtT(w16[i].ffnr, blk[i].ffnr, EE, EE, stream);
    cvtT(w16[i].ffnv, blk[i].ffnv, FF, EE, stream);
  }
  cvtT(headB, head, EE, VV, stream);

  const int nElem  = MTOK * EE;
  const int ewGrid = (nElem + 255) / 256;

  init_consts_kernel<<<NBLK, 256, 0, stream>>>(CST);
  embed_kernel<<<MTOK, 256, 0, stream>>>(H, x, emb);

  for (int i = 0; i < NBLK; ++i) {
    const float* cst = CST + i * (4*EE + 2*AA);
    const float* tmk = cst;
    const float* tmv = cst + EE;
    const float* tmr = cst + 2*EE;
    const float* cmk = cst + 3*EE;
    const float* wdk = cst + 4*EE;
    const float* u   = cst + 4*EE + AA;

    if (i == 0)
      ln_kernel<<<MTOK, 256, 0, stream>>>(H, H, blk[0].ln0g, blk[0].ln0b);

    // ---- time mix ----
    ln_kernel<<<MTOK, 256, 0, stream>>>(XLN, H, blk[i].ln1g, blk[i].ln1b);
    mix3_kernel<<<ewGrid, 256, 0, stream>>>(B1, B2, B3, XLN, tmk, tmv, tmr);
    launch_gemm(G_PLAIN, K4, B1, w16[i].attk, AA, EE, nullptr, nullptr, stream); // k
    launch_gemm(G_PLAIN, V5, B2, w16[i].attv, AA, EE, nullptr, nullptr, stream); // v
    launch_gemm(G_SIG,   R6, B3, w16[i].attr, AA, EE, nullptr, nullptr, stream); // sr
    wkv_kernel<<<(BB*AA + 255)/256, 256, 0, stream>>>(WK, K4, V5, wdk, u);
    ewmul_kernel<<<ewGrid, 256, 0, stream>>>(B1, R6, WK);                        // rw (bf16)
    launch_gemm(G_RES, H, B1, w16[i].atto, EE, AA, H, nullptr, stream);          // h += rw@Wo

    // ---- channel mix ----
    ln_kernel<<<MTOK, 256, 0, stream>>>(XLN, H, blk[i].ln2g, blk[i].ln2b);
    mix1_kernel<<<ewGrid, 256, 0, stream>>>(B1, XLN, cmk);                       // xk==xr (bf16)
    launch_gemm(G_RELUSQ_BF, FKB, B1, w16[i].ffnk, FF, EE, nullptr, nullptr, stream);
    launch_gemm(G_SIG,       R6,  B1, w16[i].ffnr, EE, EE, nullptr, nullptr, stream);
    launch_gemm(G_RESMUL,    H,  FKB, w16[i].ffnv, EE, FF, H, R6, stream);       // h += r*(k@Wv)
  }

  ln_kernel<<<MTOK, 256, 0, stream>>>(XLN, H, ln_out_g, ln_out_b);
  cvt_kernel<<<(int)(S/4 + 255)/256, 256, 0, stream>>>(XLB, XLN, (int)(S/4));
  launch_gemm(G_PLAIN, d_out, XLB, headB, VV, EE, nullptr, nullptr, stream);
}
